// PoseNetX_LIGHT_KNN_82772609729159
// MI455X (gfx1250) — matrix operand order repaired and checked by reading, not compiled, not run on hardware
//
#include <hip/hip_runtime.h>
#include <hip/hip_bf16.h>

// ---------------- problem constants ----------------
#define N_NODES 8192
#define FE_OUT  512
#define FEAT    2048
#define KNN     4
#define GSZ     512
#define NGRAPH  16
#define NEDGE   (N_NODES * KNN)   // 32768

typedef __attribute__((ext_vector_type(16))) __bf16 v16bf;
typedef __attribute__((ext_vector_type(8)))  float  v8f;

union Frag { v16bf v; uint4 u[2]; };

// ---- CDNA5 async global->LDS (ASYNCcnt path), inline asm ----
__device__ __forceinline__ void async_b128(unsigned lds_off, unsigned long long gaddr) {
    asm volatile("global_load_async_to_lds_b128 %0, %1, off"
                 :: "v"(lds_off), "v"(gaddr) : "memory");
}
__device__ __forceinline__ void wait_async_le4() {
    asm volatile("s_wait_asynccnt 4" ::: "memory");
}
__device__ __forceinline__ void wait_async_le0() {
    asm volatile("s_wait_asynccnt 0" ::: "memory");
}

// ---------------- f32 -> bf16 (flat) ----------------
__global__ void k_cvt_bf16(const float* __restrict__ in, __bf16* __restrict__ out, long n) {
    long i = (long)blockIdx.x * blockDim.x + threadIdx.x;
    if (i < n) out[i] = (__bf16)in[i];
}

// ---------------- f32 [R][C] -> bf16 [C][R] (weight transpose) ----------------
__global__ void k_transpose_bf16(const float* __restrict__ in, __bf16* __restrict__ out,
                                 int R, int C) {
    long i = (long)blockIdx.x * blockDim.x + threadIdx.x;
    if (i < (long)R * C) {
        int r = (int)(i / C), c = (int)(i % C);
        out[(long)c * R + r] = (__bf16)in[i];
    }
}

// ---------------- bf16 WMMA GEMM, async double-buffered ----------------
// C[M][N] = act( A[M][K] @ Bt[N][K]^T + bias ), Bt row-major [N][K].
// GATHER: A row for k<FEAT from glo[m], else ghi[m] (fused concat GEMM).
// OUTF32: write f32 to Cf, else bf16 to Cb. Batched via blockIdx.z.
// Block tile 128x128, K-step 32, 8 waves each computing 32x64 (2x4 WMMA frags).
template<int GATHER, int RELU, int OUTF32>
__global__ __launch_bounds__(256)
void k_gemm_bf16(const __bf16* __restrict__ A, const __bf16* __restrict__ Bt,
                 const float* __restrict__ bias,
                 float* __restrict__ Cf, __bf16* __restrict__ Cb,
                 const int* __restrict__ glo, const int* __restrict__ ghi,
                 int M, int N, int Kd,
                 long bsA, long bsB, long bsC)
{
    (void)M;
    __shared__ __bf16 As[2][128 * 40];   // 32 K + 8 pad, rows 80B (16B-aligned)
    __shared__ __bf16 Bs[2][128 * 40];

    const int t    = threadIdx.x;
    const int lane = t & 31;
    const int wid  = t >> 5;
    const int bm = blockIdx.y * 128;
    const int bn = blockIdx.x * 128;
    const int z  = blockIdx.z;

    const __bf16* Ab = A  + (long)z * bsA;
    const __bf16* Bb = Bt + (long)z * bsB;
    const long cbase = (long)z * bsC;

    // tile-fill mapping: thread -> (row 0..127, 16-elem chunk 0/16)
    const int fr  = t >> 1;
    const int fkk = (t & 1) * 16;

    // wave / lane fragment coords
    const int wm = wid >> 1;         // 0..3 -> rows wm*32
    const int wn = wid & 1;          // 0..1 -> cols wn*64
    const int fl = lane & 15;
    const int fh = lane >> 4;

    // hoisted per-thread source bases (byte addresses)
    unsigned long long aLo = 0, aHi = 0, aPtr = 0, bPtr;
    if (GATHER) {
        int e = bm + fr;
        aLo = (unsigned long long)(Ab + (long)glo[e] * FEAT);
        aHi = (unsigned long long)(Ab + (long)ghi[e] * FEAT);
    } else {
        aPtr = (unsigned long long)(Ab + (long)(bm + fr) * Kd + fkk);
    }
    bPtr = (unsigned long long)(Bb + (long)(bn + fr) * Kd + fkk);

    // LDS byte offsets (generic ptr truncates to LDS offset per aperture rule)
    const unsigned ldsA = (unsigned)(unsigned long long)&As[0][fr * 40 + fkk];
    const unsigned ldsB = (unsigned)(unsigned long long)&Bs[0][fr * 40 + fkk];
    const unsigned BUFB = (unsigned)sizeof(As[0]);   // 10240 bytes per buffer

    auto issue_tile = [&](int k0, int buf) {
        unsigned long long ga;
        if (GATHER) {
            int gk = k0 + fkk;
            ga = (gk < FEAT ? aLo : aHi) + (unsigned)((gk & (FEAT - 1)) * 2);
        } else {
            ga = aPtr + (unsigned)(k0 * 2);
        }
        unsigned la = ldsA + (unsigned)buf * BUFB;
        async_b128(la,      ga);
        async_b128(la + 16, ga + 16);
        unsigned long long gb = bPtr + (unsigned)(k0 * 2);
        unsigned lb = ldsB + (unsigned)buf * BUFB;
        async_b128(lb,      gb);
        async_b128(lb + 16, gb + 16);
    };

    v8f acc[2][4] = {};
    const int nk = Kd >> 5;

    issue_tile(0, 0);
    for (int s = 0; s < nk; ++s) {
        const int cur = s & 1;
        if (s + 1 < nk) {
            issue_tile((s + 1) * 32, cur ^ 1);   // prefetch next tile (4 asyncs)
            wait_async_le4();                    // current tile's 4 asyncs done
        } else {
            wait_async_le0();
        }
        __syncthreads();                          // cross-wave: tile visible in LDS

        Frag afr[2], bfr[4];
#pragma unroll
        for (int i = 0; i < 2; ++i) {
            const __bf16* p = &As[cur][(wm * 32 + i * 16 + fl) * 40];
            afr[i].u[0] = *(const uint4*)(p + fh * 8);        // K fh*8 .. +7
            afr[i].u[1] = *(const uint4*)(p + 16 + fh * 8);   // K 16+fh*8 .. +7
        }
#pragma unroll
        for (int j = 0; j < 4; ++j) {
            const __bf16* p = &Bs[cur][(wn * 64 + j * 16 + fl) * 40];
            bfr[j].u[0] = *(const uint4*)(p + fh * 8);        // K fh*8 .. +7
            bfr[j].u[1] = *(const uint4*)(p + 16 + fh * 8);   // K 16+fh*8 .. +7
        }
#pragma unroll
        for (int i = 0; i < 2; ++i)
#pragma unroll
            for (int j = 0; j < 4; ++j)
                acc[i][j] = __builtin_amdgcn_wmma_f32_16x16x32_bf16(
                    false, afr[i].v, false, bfr[j].v,
                    (short)0, acc[i][j], false, false);

        __syncthreads();   // all waves done reading buf `cur` before tile s+2 lands
    }

    // ---- epilogue: bias + relu, single output type ----
#pragma unroll
    for (int j = 0; j < 4; ++j) {
        int cn = bn + wn * 64 + j * 16 + fl;
        float bv = bias ? bias[cn] : 0.0f;
#pragma unroll
        for (int i = 0; i < 2; ++i) {
#pragma unroll
            for (int r = 0; r < 8; ++r) {
                int cm = bm + wm * 32 + i * 16 + fh * 8 + r;
                float v = acc[i][j][r] + bv;
                if (RELU) v = fmaxf(v, 0.0f);
                long off = cbase + (long)cm * N + cn;
                if (OUTF32) Cf[off] = v;
                else        Cb[off] = (__bf16)v;
            }
        }
    }
}

// ---------------- squared norms from Gram diagonal ----------------
__global__ void k_sqdiag(const float* __restrict__ gram, float* __restrict__ sq) {
    int i = blockIdx.x * blockDim.x + threadIdx.x;
    if (i < N_NODES) {
        int g = i >> 9, li = i & 511;
        sq[i] = gram[(long)g * GSZ * GSZ + (long)li * GSZ + li];
    }
}

// ---------------- per-node top-4 nearest neighbors ----------------
__global__ void k_knn(const float* __restrict__ gram, const float* __restrict__ sq,
                      int* __restrict__ lo, int* __restrict__ hi,
                      int* __restrict__ eidx_out)
{
    int i = blockIdx.x * blockDim.x + threadIdx.x;
    if (i >= N_NODES) return;
    int g = i >> 9, li = i & 511;
    const float* row = gram + (long)g * GSZ * GSZ + (long)li * GSZ;
    const float* sqg = sq + g * GSZ;
    float si = sq[i];
    float d0 = 3.0e38f, d1 = 3.0e38f, d2 = 3.0e38f, d3 = 3.0e38f;
    int   i0 = 0, i1 = 0, i2 = 0, i3 = 0;
    for (int j = 0; j < GSZ; ++j) {
        if (j == li) continue;
        float d = si + sqg[j] - 2.0f * row[j];
        if (d < d0)      { d3=d2;i3=i2; d2=d1;i2=i1; d1=d0;i1=i0; d0=d;i0=j; }
        else if (d < d1) { d3=d2;i3=i2; d2=d1;i2=i1; d1=d;i1=j; }
        else if (d < d2) { d3=d2;i3=i2; d2=d;i2=j; }
        else if (d < d3) { d3=d;i3=j; }
    }
    int nb[4] = { i0, i1, i2, i3 };
#pragma unroll
    for (int t = 0; t < 4; ++t) {
        int e   = i * KNN + t;
        int src = g * GSZ + nb[t];
        int dst = i;
        lo[e] = src < dst ? src : dst;
        hi[e] = src > dst ? src : dst;
        eidx_out[e]         = src;   // edge_index[0] = neighbor
        eidx_out[NEDGE + e] = dst;   // edge_index[1] = node
    }
}

// ---------------- two 2048->3 heads fused, writes [M][6] f32 ----------------
__global__ void k_head6(const __bf16* __restrict__ X, int Kd,
                        const float* __restrict__ w1, const float* __restrict__ b1,
                        const float* __restrict__ w2, const float* __restrict__ b2,
                        float* __restrict__ out, int M)
{
    int tid = blockIdx.x * blockDim.x + threadIdx.x;
    if (tid >= M * 3) return;
    int i = tid / 3, c = tid - i * 3;
    const __bf16* x = X + (long)i * Kd;
    float a1 = 0.f, a2 = 0.f;
    for (int k = 0; k < Kd; ++k) {
        float xv = (float)x[k];
        a1 += xv * w1[k * 3 + c];
        a2 += xv * w2[k * 3 + c];
    }
    out[(long)i * 6 + c]     = a1 + b1[c];
    out[(long)i * 6 + 3 + c] = a2 + b2[c];
}

// ---------------- launcher ----------------
extern "C" void kernel_launch(void* const* d_in, const int* in_sizes, int n_in,
                              void* d_out, int out_size, void* d_ws, size_t ws_size,
                              hipStream_t stream)
{
    (void)in_sizes; (void)n_in; (void)out_size; (void)ws_size;
    const float* x_feat  = (const float*)d_in[0];
    /* d_in[1] = batch (structure known: 16 graphs of 512) */
    const float* fc_w    = (const float*)d_in[2];
    const float* fc_b    = (const float*)d_in[3];
    const float* proj_w  = (const float*)d_in[4];
    const float* proj_b  = (const float*)d_in[5];
    const float* m1_w    = (const float*)d_in[6];
    const float* m1_b    = (const float*)d_in[7];
    const float* m2_w    = (const float*)d_in[8];
    const float* m2_b    = (const float*)d_in[9];
    const float* xyz_w   = (const float*)d_in[10];
    const float* xyz_b   = (const float*)d_in[11];
    const float* wpqr_w  = (const float*)d_in[12];
    const float* wpqr_b  = (const float*)d_in[13];
    const float* xyzR_w  = (const float*)d_in[14];
    const float* xyzR_b  = (const float*)d_in[15];
    const float* wpqrR_w = (const float*)d_in[16];
    const float* wpqrR_b = (const float*)d_in[17];

    // ---- workspace carve-out (256B aligned) ----
    char* p = (char*)d_ws;
    auto take = [&](size_t bytes) { char* r = p; p += (bytes + 255) & ~(size_t)255; return (void*)r; };
    __bf16* xfeat_bf = (__bf16*)take((size_t)N_NODES * FE_OUT * 2);
    __bf16* fcwT     = (__bf16*)take((size_t)FEAT * FE_OUT * 2);        // [2048][512]
    __bf16* projwT   = (__bf16*)take((size_t)FEAT * (2 * FEAT) * 2);    // [2048][4096]
    __bf16* m1T      = (__bf16*)take((size_t)FEAT * FEAT * 2);
    __bf16* m2T      = (__bf16*)take((size_t)FEAT * FEAT * 2);
    __bf16* x_bf     = (__bf16*)take((size_t)N_NODES * FEAT * 2);
    float*  gram     = (float*)take((size_t)NGRAPH * GSZ * GSZ * 4);
    float*  sq       = (float*)take((size_t)N_NODES * 4);
    int*    lo       = (int*)take((size_t)NEDGE * 4);
    int*    hi       = (int*)take((size_t)NEDGE * 4);
    __bf16* ef_a     = (__bf16*)take((size_t)NEDGE * FEAT * 2);
    __bf16* ef_b     = (__bf16*)take((size_t)NEDGE * FEAT * 2);

    // ---- precision conversions / weight transposes ----
    {
        long n = (long)N_NODES * FE_OUT;
        k_cvt_bf16<<<(unsigned)((n + 255) / 256), 256, 0, stream>>>(x_feat, xfeat_bf, n);
    }
    k_transpose_bf16<<<(unsigned)(((long)FE_OUT * FEAT + 255) / 256), 256, 0, stream>>>(fc_w, fcwT, FE_OUT, FEAT);
    k_transpose_bf16<<<(unsigned)(((long)2 * FEAT * FEAT + 255) / 256), 256, 0, stream>>>(proj_w, projwT, 2 * FEAT, FEAT);
    k_transpose_bf16<<<(unsigned)(((long)FEAT * FEAT + 255) / 256), 256, 0, stream>>>(m1_w, m1T, FEAT, FEAT);
    k_transpose_bf16<<<(unsigned)(((long)FEAT * FEAT + 255) / 256), 256, 0, stream>>>(m2_w, m2T, FEAT, FEAT);

    dim3 blk(256);
    // 1) x = x_feat @ fc_w + fc_b               [8192 x 2048] -> bf16
    k_gemm_bf16<0,0,0><<<dim3(FEAT / 128, N_NODES / 128, 1), blk, 0, stream>>>(
        xfeat_bf, fcwT, fc_b, nullptr, x_bf, nullptr, nullptr,
        N_NODES, FEAT, FE_OUT, 0, 0, 0);
    // 2) per-graph Gram G = Xg @ Xg^T           16 x [512 x 512] -> f32
    k_gemm_bf16<0,0,1><<<dim3(GSZ / 128, GSZ / 128, NGRAPH), blk, 0, stream>>>(
        x_bf, x_bf, nullptr, gram, nullptr, nullptr, nullptr,
        GSZ, GSZ, FEAT,
        (long)GSZ * FEAT, (long)GSZ * FEAT, (long)GSZ * GSZ);
    // 3) knn top-4
    k_sqdiag<<<(N_NODES + 255) / 256, 256, 0, stream>>>(gram, sq);
    int* eidx_out = (int*)((float*)d_out + (size_t)N_NODES * 6 + (size_t)NEDGE * 6);
    k_knn<<<(N_NODES + 255) / 256, 256, 0, stream>>>(gram, sq, lo, hi, eidx_out);
    // 4) proj: relu( cat(x[lo],x[hi]) @ proj_w + b )   [32768 x 2048], gather GEMM
    k_gemm_bf16<1,1,0><<<dim3(FEAT / 128, NEDGE / 128, 1), blk, 0, stream>>>(
        x_bf, projwT, proj_b, nullptr, ef_a, lo, hi,
        NEDGE, FEAT, 2 * FEAT, 0, 0, 0);
    // 5) mlp1 (relu)
    k_gemm_bf16<0,1,0><<<dim3(FEAT / 128, NEDGE / 128, 1), blk, 0, stream>>>(
        ef_a, m1T, m1_b, nullptr, ef_b, nullptr, nullptr,
        NEDGE, FEAT, FEAT, 0, 0, 0);
    // 6) mlp2 (no relu)
    k_gemm_bf16<0,0,0><<<dim3(FEAT / 128, NEDGE / 128, 1), blk, 0, stream>>>(
        ef_b, m2T, m2_b, nullptr, ef_a, nullptr, nullptr,
        NEDGE, FEAT, FEAT, 0, 0, 0);
    // 7) heads
    float* node_out = (float*)d_out;
    float* edge_out = (float*)d_out + (size_t)N_NODES * 6;
    k_head6<<<(N_NODES * 3 + 255) / 256, 256, 0, stream>>>(
        x_bf, FEAT, xyz_w, xyz_b, wpqr_w, wpqr_b, node_out, N_NODES);
    k_head6<<<(NEDGE * 3 + 255) / 256, 256, 0, stream>>>(
        ef_a, FEAT, xyzR_w, xyzR_b, wpqrR_w, wpqrR_b, edge_out, NEDGE);
}
